// MultiHeadAttn_5592047419551
// MI455X (gfx1250) — matrix operand order repaired and checked by reading, not compiled, not run on hardware
//
#include <hip/hip_runtime.h>

typedef __attribute__((ext_vector_type(16))) __bf16 v16bf;
typedef __attribute__((ext_vector_type(8)))  float  v8f;

union Frag  { uint4 q[2]; v16bf b; };        // 16 bf16 per lane = 8 VGPRs
union PK4   { uint2 u; __bf16 e[4]; };
union PK2   { unsigned int u; __bf16 e[2]; };
union V8U   { uint4 q; __bf16 e[8]; };

#define BSZ   2
#define NSEQ  4096
#define DMOD  512
#define NH    8
#define DH    64
#define CTX   10

// ---------------- stage 0a: x f32 -> bf16 (packed) ----------------
__global__ __launch_bounds__(256)
void convert_x(const float* __restrict__ x, __bf16* __restrict__ xb) {
    int gid = blockIdx.x * blockDim.x + threadIdx.x;      // one thread : 4 elems
    size_t i4 = (size_t)gid * 4;
    float4 f = *(const float4*)(x + i4);
    PK4 p;
    p.e[0] = (__bf16)f.x; p.e[1] = (__bf16)f.y;
    p.e[2] = (__bf16)f.z; p.e[3] = (__bf16)f.w;
    *(uint2*)(xb + i4) = p.u;
}

// ---------------- stage 0b: weights f32 -> bf16, transposed (N-major) ---------
// WqT/WvT: [H][DH][DMOD], WupT: [DMOD cols][DMOD k]
__global__ __launch_bounds__(256)
void convert_wT(const float* __restrict__ Wq, const float* __restrict__ Wv,
                const float* __restrict__ Wup,
                __bf16* __restrict__ WqT, __bf16* __restrict__ WvT,
                __bf16* __restrict__ WupT) {
    int o = blockIdx.x * blockDim.x + threadIdx.x;        // 262144 threads
    // per-head projections
    int k = o & (DMOD - 1);
    int n = (o >> 9) & (DH - 1);
    int h = o >> 15;
    size_t src = ((size_t)h * DMOD + k) * DH + n;
    WqT[o] = (__bf16)Wq[src];
    WvT[o] = (__bf16)Wv[src];
    // up-projection: o = n*512 + k  ->  Wup[k][n]
    int nu = o >> 9;
    WupT[o] = (__bf16)Wup[(size_t)k * DMOD + nu];
}

// WMMA fragment loaders: operand rows are contiguous bf16 in global memory.
// A (16x32, M-major): lane m = lane&15, half = lane>>4; chunks at k = 8*half and 16+8*half
__device__ __forceinline__ Frag load_a_frag(const __bf16* rowPtr, int half) {
    Frag f;
    const __bf16* p = rowPtr + half * 8;
    f.q[0] = *(const uint4*)p;
    f.q[1] = *(const uint4*)(p + 16);
    return f;
}
// B (32x16, N-major/transposed): lane n = lane&15; chunks at k = 8*half and 16+8*half
__device__ __forceinline__ Frag load_b_frag(const __bf16* rowPtr, int half) {
    Frag f;
    const __bf16* p = rowPtr + half * 8;
    f.q[0] = *(const uint4*)p;
    f.q[1] = *(const uint4*)(p + 16);
    return f;
}

// ---------------- stage 1: Q,V projections (LDS-free WMMA) ----------------
// 8 waves/block; waves 0-3 -> Q, 4-7 -> V; each wave: 64 rows x 64 cols
__global__ __launch_bounds__(256)
void proj_qv(const __bf16* __restrict__ xb,
             const __bf16* __restrict__ WqT,
             const __bf16* __restrict__ WvT,
             __bf16* __restrict__ Qb,
             __bf16* __restrict__ Vb) {
    const int tid  = threadIdx.x;
    const int wave = tid >> 5, lane = tid & 31;
    const int b = blockIdx.z, h = blockIdx.y;
    const int isV   = wave >> 2;
    const int rBase = blockIdx.x * 256 + (wave & 3) * 64;
    const int half  = lane >> 4, ml = lane & 15;

    const __bf16* A0 = xb + (size_t)b * NSEQ * DMOD;
    const __bf16* WT = (isV ? WvT : WqT) + (size_t)h * DH * DMOD;

    v8f acc[4][4] = {};
#pragma unroll 1
    for (int kb = 0; kb < DMOD; kb += 32) {
        Frag af[4];
#pragma unroll
        for (int s = 0; s < 4; ++s) {
            const __bf16* pa = A0 + (size_t)(rBase + s * 16 + ml) * DMOD + kb;
            af[s] = load_a_frag(pa, half);
            __builtin_prefetch(pa + 32, 0, 1);            // next K slice
        }
#pragma unroll
        for (int c = 0; c < 4; ++c) {
            Frag bfg = load_b_frag(WT + (size_t)(c * 16 + ml) * DMOD + kb, half);
#pragma unroll
            for (int s = 0; s < 4; ++s)
                acc[s][c] = __builtin_amdgcn_wmma_f32_16x16x32_bf16(
                    false, af[s].b, false, bfg.b, (short)0, acc[s][c], false, false);
        }
    }

    __bf16* Out = (isV ? Vb : Qb) + (size_t)(b * NH + h) * NSEQ * DH;
#pragma unroll
    for (int s = 0; s < 4; ++s)
#pragma unroll
        for (int c = 0; c < 4; ++c)
#pragma unroll
            for (int r = 0; r < 8; ++r) {
                int row = rBase + s * 16 + half * 8 + r;
                int col = c * 16 + ml;
                Out[(size_t)row * DH + col] = (__bf16)acc[s][c][r];
            }
}

// ---------------- stage 2: banded softmax attention ----------------
__global__ __launch_bounds__(256)
void band_attn(const __bf16* __restrict__ Qb,
               const __bf16* __restrict__ Vb,
               __bf16* __restrict__ Cb) {
    const int tid  = threadIdx.x;
    const int lane = tid & 31;
    const int wid  = blockIdx.x * 8 + (tid >> 5);
    const int b    = wid / (NH * NSEQ);
    const int rem  = wid % (NH * NSEQ);
    const int h    = rem / NSEQ;
    const int i    = rem % NSEQ;

    const __bf16* Qrow = Qb + ((size_t)(b * NH + h) * NSEQ + i) * DH;
    const __bf16* Vmat = Vb +  (size_t)(b * NH + h) * NSEQ * DH;

    const int d = lane - CTX;
    const int j = i + d;
    const bool valid = (lane <= 2 * CTX) && (d != 0) && (j >= 0) && (j < NSEQ);

    float s = -1e30f;
    if (valid) {
        const __bf16* Vrow = Vmat + (size_t)j * DH;
        float dot = 0.f;
#pragma unroll
        for (int t = 0; t < 8; ++t) {
            V8U qq, vv;
            qq.q = *(const uint4*)(Qrow + t * 8);
            vv.q = *(const uint4*)(Vrow + t * 8);
#pragma unroll
            for (int u = 0; u < 8; ++u) dot += (float)qq.e[u] * (float)vv.e[u];
        }
        s = dot * (1.0f / 16.0f);
    }
    float m = s;
#pragma unroll
    for (int off = 16; off > 0; off >>= 1) m = fmaxf(m, __shfl_xor(m, off, 32));
    float e = valid ? __expf(s - m) : 0.f;
    float sum = e;
#pragma unroll
    for (int off = 16; off > 0; off >>= 1) sum += __shfl_xor(sum, off, 32);
    const float w = e / sum;

    const int k0 = lane * 2;
    float a0 = 0.f, a1 = 0.f;
#pragma unroll
    for (int jj = 0; jj <= 2 * CTX; ++jj) {
        float wj = __shfl(w, jj, 32);                     // 0 for invalid lanes
        int jn = i + jj - CTX;
        if (jn >= 0 && jn < NSEQ) {
            PK2 v2;
            v2.u = *(const unsigned int*)(Vmat + (size_t)jn * DH + k0);
            a0 += wj * (float)v2.e[0];
            a1 += wj * (float)v2.e[1];
        }
    }
    PK2 o;
    o.e[0] = (__bf16)a0; o.e[1] = (__bf16)a1;
    *(unsigned int*)(Cb + ((size_t)b * NSEQ + i) * (NH * DH) + h * DH + k0) = o.u;
}

// ---------------- stage 3: heads @ Wup -> f32 out (LDS-free WMMA) -------------
// 8 waves/block; block tile 256 rows x 128 cols; wave: 64 rows x 64 cols
__global__ __launch_bounds__(256)
void out_proj(const __bf16* __restrict__ Cb,
              const __bf16* __restrict__ WupT,
              float* __restrict__ out) {
    const int tid  = threadIdx.x;
    const int wave = tid >> 5, lane = tid & 31;
    const int rBase = blockIdx.x * 256 + (wave & 3) * 64;
    const int cBase = blockIdx.y * 128 + (wave >> 2) * 64;
    const int half  = lane >> 4, ml = lane & 15;

    v8f acc[4][4] = {};
#pragma unroll 1
    for (int kb = 0; kb < DMOD; kb += 32) {
        Frag af[4];
#pragma unroll
        for (int s = 0; s < 4; ++s) {
            const __bf16* pa = Cb + (size_t)(rBase + s * 16 + ml) * DMOD + kb;
            af[s] = load_a_frag(pa, half);
            __builtin_prefetch(pa + 32, 0, 1);
        }
#pragma unroll
        for (int c = 0; c < 4; ++c) {
            Frag bfg = load_b_frag(WupT + (size_t)(cBase + c * 16 + ml) * DMOD + kb, half);
#pragma unroll
            for (int s = 0; s < 4; ++s)
                acc[s][c] = __builtin_amdgcn_wmma_f32_16x16x32_bf16(
                    false, af[s].b, false, bfg.b, (short)0, acc[s][c], false, false);
        }
    }

#pragma unroll
    for (int s = 0; s < 4; ++s)
#pragma unroll
        for (int c = 0; c < 4; ++c)
#pragma unroll
            for (int r = 0; r < 8; ++r) {
                int row = rBase + s * 16 + half * 8 + r;
                int col = cBase + c * 16 + ml;
                out[(size_t)row * DMOD + col] = acc[s][c][r];
            }
}

extern "C" void kernel_launch(void* const* d_in, const int* in_sizes, int n_in,
                              void* d_out, int out_size, void* d_ws, size_t ws_size,
                              hipStream_t stream) {
    const float* x   = (const float*)d_in[0];
    const float* Wq  = (const float*)d_in[1];
    const float* Wv  = (const float*)d_in[2];
    const float* Wup = (const float*)d_in[3];
    float* out = (float*)d_out;

    __bf16* ws   = (__bf16*)d_ws;
    __bf16* xb   = ws;                        // B*N*D      = 4194304
    __bf16* WqT  = xb   + 4194304;            // H*DH*D     =  262144
    __bf16* WvT  = WqT  + 262144;
    __bf16* WupT = WvT  + 262144;
    __bf16* Qb   = WupT + 262144;             // B*H*N*DH   = 4194304
    __bf16* Vb   = Qb   + 4194304;
    __bf16* Cb   = Vb   + 4194304;            // B*N*H*DH   = 4194304
    // total ~35.1 MB of workspace

    convert_x <<<(BSZ * NSEQ * DMOD) / 4 / 256, 256, 0, stream>>>(x, xb);
    convert_wT<<<(NH * DMOD * DH) / 256, 256, 0, stream>>>(Wq, Wv, Wup, WqT, WvT, WupT);

    dim3 g1(NSEQ / 256, NH, BSZ);             // (16, 8, 2)
    proj_qv<<<g1, 256, 0, stream>>>(xb, WqT, WvT, Qb, Vb);

    band_attn<<<(BSZ * NH * NSEQ) / 8, 256, 0, stream>>>(Qb, Vb, Cb);

    dim3 g3((BSZ * NSEQ) / 256, DMOD / 128);  // (32, 4)
    out_proj<<<g3, 256, 0, stream>>>(Cb, WupT, out);
}
